// mixprop_gated_attention_66786741452946
// MI455X (gfx1250) — matrix operand order repaired and checked by reading, not compiled, not run on hardware
//
#include <hip/hip_runtime.h>

// ---------------------------------------------------------------------------
// mixprop gated attention, CDNA5 (gfx1250, wave32)
// Shapes: N=32, C=32, V=2048, L=12.  M = N*C*L = 12288.
// Main GEMM: diffT[m,w] = sum_v hT[m,v] * adj[v,w]
//   - bf16 hi/lo split (3 WMMAs: hi*hi + hi*lo + lo*hi, ~fp32 accuracy)
//   - A pre-split into bf16 (natural k order), staged to LDS with
//     global_load_async_to_lds_b128 (double-buffered, ASYNCcnt)
//   - B = adj transposed/split, resident in 192MB L2
// ---------------------------------------------------------------------------

typedef __bf16 bf16;
typedef __attribute__((ext_vector_type(16))) bf16  v16bf;
typedef __attribute__((ext_vector_type(8)))  bf16  v8bf;
typedef __attribute__((ext_vector_type(8)))  float v8f;

#define NB   32
#define CB   32
#define VB   2048
#define LB   12
#define MB   (NB * CB * LB)   // 12288

// -------------------------- row sums of W_a/W_b/W_o -------------------------
__global__ __launch_bounds__(96) void prep_vec_kernel(
    const float* __restrict__ Wa, const float* __restrict__ Wb,
    const float* __restrict__ Wo, float* __restrict__ w3) {
  int t = threadIdx.x;           // 0..95
  const float* W = (t < 32) ? Wa : ((t < 64) ? Wb : Wo);
  int c = t & 31;
  float s = 0.f;
  #pragma unroll
  for (int j = 0; j < 32; ++j) s += W[c * 32 + j];
  w3[t] = s;                     // [0,32)=wa  [32,64)=wb  [64,96)=wo
}

// ------------------- h_in (N,C,V,L) -> hT[(n,c,l), v] -----------------------
__global__ __launch_bounds__(256) void transpose_in_kernel(
    const float* __restrict__ h, float* __restrict__ hT) {
  int t  = threadIdx.x;
  int nc = blockIdx.y;
  int v0 = blockIdx.x * 256;
  const float* src = h + (size_t)nc * (VB * LB) + (size_t)v0 * LB + (size_t)t * LB;
  float val[LB];
  #pragma unroll
  for (int i = 0; i < LB; ++i) val[i] = src[i];
  #pragma unroll
  for (int l = 0; l < LB; ++l)
    hT[((size_t)nc * LB + l) * VB + v0 + t] = val[l];
}

// ---------------- outT[(n,c,l), v] -> d_out (N,C,V,L) -----------------------
__global__ __launch_bounds__(256) void transpose_out_kernel(
    const float* __restrict__ outT, float* __restrict__ dst) {
  int t  = threadIdx.x;
  int nc = blockIdx.y;
  int v0 = blockIdx.x * 256;
  float val[LB];
  #pragma unroll
  for (int l = 0; l < LB; ++l)
    val[l] = outT[((size_t)nc * LB + l) * VB + v0 + t];
  float* d = dst + (size_t)nc * (VB * LB) + (size_t)v0 * LB + (size_t)t * LB;
  #pragma unroll
  for (int i = 0; i < LB; ++i) d[i] = val[i];
}

// --------- adj (V,V) -> transposed bf16 hi/lo: Bt[w*V + v] ------------------
__global__ __launch_bounds__(256) void prep_adj_kernel(
    const float* __restrict__ adj, bf16* __restrict__ Bhi, bf16* __restrict__ Blo) {
  __shared__ float tile[32][33];
  int tx = threadIdx.x & 31;
  int ty = threadIdx.x >> 5;
  int v0 = blockIdx.y * 32;
  int w0 = blockIdx.x * 32;
  #pragma unroll
  for (int j = 0; j < 4; ++j)
    tile[ty + 8 * j][tx] = adj[(size_t)(v0 + ty + 8 * j) * VB + (w0 + tx)];
  __syncthreads();
  #pragma unroll
  for (int j = 0; j < 4; ++j) {
    int wl = ty + 8 * j, vl = tx;
    float x = tile[vl][wl];
    bf16 hi = (bf16)x;
    size_t o = (size_t)(w0 + wl) * VB + (v0 + vl);
    Bhi[o] = hi;
    Blo[o] = (bf16)(x - (float)hi);
  }
}

// ------ hT fp32 -> At_hi/At_lo bf16, natural k order within each K=32 -----
// Within each 32-K block, 8-elem groups are stored in natural order so that
// lane group g reads 16 contiguous bf16 = {K g*16..g*16+15}.
__global__ __launch_bounds__(256) void split_kernel(
    const float* __restrict__ hT, bf16* __restrict__ Ahi, bf16* __restrict__ Alo) {
  size_t i = ((size_t)blockIdx.x * 256 + threadIdx.x) * 8;   // source element idx
  v8f x = *(const v8f*)(hT + i);
  size_t g  = (i >> 3) & 3;                                  // 8-group within 32
  size_t gs = g;                                             // natural order
  size_t dst = (i & ~(size_t)31) + gs * 8;
  v8bf hi, lo;
  #pragma unroll
  for (int j = 0; j < 8; ++j) {
    float f = x[j];
    bf16 h = (bf16)f;
    hi[j] = h;
    lo[j] = (bf16)(f - (float)h);
  }
  *(v8bf*)(Ahi + dst) = hi;
  *(v8bf*)(Alo + dst) = lo;
}

// ---------------------------- split-bf16 WMMA GEMM --------------------------
// block = 256 thr (8 waves). M-block 64 (4 m-tiles), W-block 256 (2 wt/wave).
// A tile (64x32 hi+lo = 8KB) async-staged to LDS, double buffered.
__global__ __launch_bounds__(256) void gemm_kernel(
    const bf16* __restrict__ Ahi, const bf16* __restrict__ Alo,
    const bf16* __restrict__ Bhi, const bf16* __restrict__ Blo,
    float* __restrict__ out) {
  __shared__ alignas(128) bf16 sbuf[2][2][64 * 32];   // [buf][hi/lo][row*32+pos]

  const int tid  = threadIdx.x;
  const int lane = tid & 31;
  const int wv   = tid >> 5;
  const int m0   = blockIdx.y * 64;
  const int lh   = lane & 15;
  const int g16  = (lane >> 4) << 4;    // 0 / 16 (elements)
  const int roff = (lane >> 4) << 3;    // D rows: 0 / 8
  const int w0   = blockIdx.x * 256 + wv * 32;       // this wave: cols [w0, w0+32)

  // staging role: thread t copies 16B chunk (row = t>>2, part = t&3)
  const int srow  = tid >> 2;
  const int spart = tid & 3;
  const size_t abase = (size_t)(m0 + srow) * VB + spart * 8;  // + k0 at issue
  const int    soff  = srow * 32 + spart * 8;                 // LDS element offset

  // B pointers (adjT rows, contiguous K per lane half)
  const bf16* bp_h0 = Bhi + (size_t)(w0 + lh) * VB + g16;
  const bf16* bp_l0 = Blo + (size_t)(w0 + lh) * VB + g16;
  const bf16* bp_h1 = bp_h0 + (size_t)16 * VB;
  const bf16* bp_l1 = bp_l0 + (size_t)16 * VB;

  const v8f vzero = {0.f, 0.f, 0.f, 0.f, 0.f, 0.f, 0.f, 0.f};
  v8f acc[4][2];
  #pragma unroll
  for (int mt = 0; mt < 4; ++mt) { acc[mt][0] = vzero; acc[mt][1] = vzero; }

#define STAGE_A(buf, kk)                                                        \
  do {                                                                          \
    asm volatile("global_load_async_to_lds_b128 %0, %1, off"                    \
                 :: "v"((unsigned)(uintptr_t)&sbuf[(buf)][0][soff]),            \
                    "v"(Ahi + abase + (kk)) : "memory");                        \
    asm volatile("global_load_async_to_lds_b128 %0, %1, off"                    \
                 :: "v"((unsigned)(uintptr_t)&sbuf[(buf)][1][soff]),            \
                    "v"(Alo + abase + (kk)) : "memory");                        \
  } while (0)

  STAGE_A(0, 0);

  for (int ki = 0; ki < VB / 32; ++ki) {
    const int b  = ki & 1;
    const int k0 = ki * 32;
    asm volatile("s_wait_asynccnt 0x0" ::: "memory");  // own chunks landed
    __syncthreads();                                   // all chunks landed; prev reads done
    if (ki + 1 < VB / 32) STAGE_A(b ^ 1, k0 + 32);     // prefetch next buffer

    v16bf bh0 = *(const v16bf*)(bp_h0 + k0);
    v16bf bl0 = *(const v16bf*)(bp_l0 + k0);
    v16bf bh1 = *(const v16bf*)(bp_h1 + k0);
    v16bf bl1 = *(const v16bf*)(bp_l1 + k0);

    #pragma unroll
    for (int mt = 0; mt < 4; ++mt) {
      v16bf ah = *(const v16bf*)&sbuf[b][0][(mt * 16 + lh) * 32 + g16];
      v16bf al = *(const v16bf*)&sbuf[b][1][(mt * 16 + lh) * 32 + g16];
      acc[mt][0] = __builtin_amdgcn_wmma_f32_16x16x32_bf16(false, ah, false, bh0,
                                                           (short)0, acc[mt][0], false, false);
      acc[mt][0] = __builtin_amdgcn_wmma_f32_16x16x32_bf16(false, ah, false, bl0,
                                                           (short)0, acc[mt][0], false, false);
      acc[mt][0] = __builtin_amdgcn_wmma_f32_16x16x32_bf16(false, al, false, bh0,
                                                           (short)0, acc[mt][0], false, false);
      acc[mt][1] = __builtin_amdgcn_wmma_f32_16x16x32_bf16(false, ah, false, bh1,
                                                           (short)0, acc[mt][1], false, false);
      acc[mt][1] = __builtin_amdgcn_wmma_f32_16x16x32_bf16(false, ah, false, bl1,
                                                           (short)0, acc[mt][1], false, false);
      acc[mt][1] = __builtin_amdgcn_wmma_f32_16x16x32_bf16(false, al, false, bh1,
                                                           (short)0, acc[mt][1], false, false);
    }
  }
#undef STAGE_A

  // D layout: lanes 0-15: (M=r, N=lane); lanes 16-31: (M=8+r, N=lane-16)
  #pragma unroll
  for (int mt = 0; mt < 4; ++mt) {
    #pragma unroll
    for (int r = 0; r < 8; ++r) {
      size_t row = (size_t)(m0 + mt * 16 + roff + r) * VB;
      out[row + w0 + lh]      = acc[mt][0][r];
      out[row + w0 + 16 + lh] = acc[mt][1][r];
    }
  }
}

// ----------------- alpha-gate + mix + 1x1 conv + ReLU -----------------------
__global__ __launch_bounds__(256) void mix_kernel(
    const float* __restrict__ hT_in, const float* __restrict__ hT_cur,
    const float* __restrict__ diffT, const float* __restrict__ w3,
    const float* __restrict__ uvec, const float* __restrict__ mlp_w,
    const float* __restrict__ mlp_b, float* __restrict__ hT_next) {
  __shared__ float s_w[1024];
  __shared__ float s_u[32], s_wa[32], s_wb[32], s_b[32];
  int t = threadIdx.x;
  for (int i = t; i < 1024; i += 256) s_w[i] = mlp_w[i];
  if (t < 32) { s_u[t] = uvec[t]; s_wa[t] = w3[t]; s_wb[t] = w3[32 + t]; s_b[t] = mlp_b[t]; }
  __syncthreads();

  int nl = blockIdx.y;                // n*12 + l
  int n  = nl / LB, l = nl % LB;
  int w  = blockIdx.x * 256 + t;
  size_t base = ((size_t)n * (CB * LB) + l) * VB + w;

  float s_l = 0.f, s_in = 0.f;
  float hin[CB], dif[CB];
  #pragma unroll
  for (int c = 0; c < CB; ++c) {
    size_t idx = base + (size_t)c * (LB * VB);
    s_l += hT_cur[idx] * s_wb[c];
    float hv = hT_in[idx];
    hin[c] = hv;  s_in += hv * s_wa[c];
    dif[c] = diffT[idx];
  }
  float mx = -3.0e38f;
  #pragma unroll
  for (int c = 0; c < CB; ++c) {
    float a = s_u[c] * s_l, bb = s_u[c] * s_in;
    mx = fmaxf(mx, fmaxf(a, bb));
  }
  float Z = 0.f;
  #pragma unroll
  for (int c = 0; c < CB; ++c)
    Z += __expf(s_u[c] * s_l - mx) + __expf(s_u[c] * s_in - mx);
  float alpha = __expf(s_u[0] * s_l - mx) / Z;

  #pragma unroll
  for (int c = 0; c < CB; ++c) dif[c] = (1.f - alpha) * hin[c] + alpha * dif[c];

  #pragma unroll
  for (int o = 0; o < CB; ++o) {
    float acc = s_b[o];
    #pragma unroll
    for (int c = 0; c < CB; ++c) acc += s_w[o * 32 + c] * dif[c];
    hT_next[base + (size_t)o * (LB * VB)] = fmaxf(acc, 0.f);
  }
}

// -------------------- depth attention over {h_in, h1, h2} -------------------
__global__ __launch_bounds__(256) void final_kernel(
    const float* __restrict__ h0, const float* __restrict__ h1,
    const float* __restrict__ h2, const float* __restrict__ w3,
    const float* __restrict__ vvec, float* __restrict__ outT) {
  __shared__ float s_v[32], s_wo[32];
  int t = threadIdx.x;
  if (t < 32) { s_v[t] = vvec[t]; s_wo[t] = w3[64 + t]; }
  __syncthreads();

  int nl = blockIdx.y;
  int n  = nl / LB, l = nl % LB;
  int w  = blockIdx.x * 256 + t;
  size_t base = ((size_t)n * (CB * LB) + l) * VB + w;

  float s0 = 0.f, s1 = 0.f, s2 = 0.f;
  #pragma unroll
  for (int c = 0; c < CB; ++c) {
    size_t idx = base + (size_t)c * (LB * VB);
    s0 += h0[idx] * s_wo[c];
    s1 += h1[idx] * s_wo[c];
    s2 += h2[idx] * s_wo[c];
  }
  float mx = -3.0e38f;
  #pragma unroll
  for (int c = 0; c < CB; ++c) {
    float vc = s_v[c];
    mx = fmaxf(mx, fmaxf(vc * s0, fmaxf(vc * s1, vc * s2)));
  }
  float Z = 0.f;
  #pragma unroll
  for (int c = 0; c < CB; ++c) {
    float vc = s_v[c];
    Z += __expf(vc * s0 - mx) + __expf(vc * s1 - mx) + __expf(vc * s2 - mx);
  }
  float invZ = 1.f / Z;
  #pragma unroll
  for (int c = 0; c < CB; ++c) {
    size_t idx = base + (size_t)c * (LB * VB);
    float vc = s_v[c];
    float e0 = __expf(vc * s0 - mx);
    float e1 = __expf(vc * s1 - mx);
    float e2 = __expf(vc * s2 - mx);
    outT[idx] = (e0 * h0[idx] + e1 * h1[idx] + e2 * h2[idx]) * invZ;
  }
}

// ---------------------------------------------------------------------------
extern "C" void kernel_launch(void* const* d_in, const int* in_sizes, int n_in,
                              void* d_out, int out_size, void* d_ws, size_t ws_size,
                              hipStream_t stream) {
  const float* h_in  = (const float*)d_in[0];
  const float* adj   = (const float*)d_in[1];
  const float* W_a   = (const float*)d_in[2];
  const float* W_b   = (const float*)d_in[3];
  const float* W_o   = (const float*)d_in[4];
  const float* v_vec = (const float*)d_in[5];
  const float* u_vec = (const float*)d_in[6];
  const float* mlp_w = (const float*)d_in[7];
  const float* mlp_b = (const float*)d_in[8];
  float* out = (float*)d_out;

  char* ws = (char*)d_ws;
  const size_t sz  = (size_t)MB * VB * sizeof(float);   // 100,663,296 B
  const size_t szA = (size_t)MB * VB * sizeof(bf16);    //  50,331,648 B
  const size_t szB = (size_t)VB * VB * sizeof(bf16);    //   8,388,608 B
  float* hT0   = (float*)(ws);
  float* hT1   = (float*)(ws + sz);
  float* hT2   = (float*)(ws + 2 * sz);
  float* diffT = (float*)(ws + 3 * sz);                 // reused as outT
  bf16*  Ahi   = (bf16*)(ws + 4 * sz);
  bf16*  Alo   = (bf16*)(ws + 4 * sz + szA);
  bf16*  Bhi   = (bf16*)(ws + 4 * sz + 2 * szA);
  bf16*  Blo   = (bf16*)(ws + 4 * sz + 2 * szA + szB);
  float* w3    = (float*)(ws + 4 * sz + 2 * szA + 2 * szB);

  prep_vec_kernel<<<1, 96, 0, stream>>>(W_a, W_b, W_o, w3);
  transpose_in_kernel<<<dim3(VB / 256, NB * CB), 256, 0, stream>>>(h_in, hT0);
  prep_adj_kernel<<<dim3(VB / 32, VB / 32), 256, 0, stream>>>(adj, Bhi, Blo);

  const int splitBlocks = (int)((size_t)MB * VB / 8 / 256);   // 12288
  // step 0
  split_kernel<<<splitBlocks, 256, 0, stream>>>(hT0, Ahi, Alo);
  gemm_kernel<<<dim3(VB / 256, MB / 64), 256, 0, stream>>>(Ahi, Alo, Bhi, Blo, diffT);
  mix_kernel<<<dim3(VB / 256, NB * LB), 256, 0, stream>>>(hT0, hT0, diffT, w3,
                                                          u_vec, mlp_w, mlp_b, hT1);
  // step 1
  split_kernel<<<splitBlocks, 256, 0, stream>>>(hT1, Ahi, Alo);
  gemm_kernel<<<dim3(VB / 256, MB / 64), 256, 0, stream>>>(Ahi, Alo, Bhi, Blo, diffT);
  mix_kernel<<<dim3(VB / 256, NB * LB), 256, 0, stream>>>(hT0, hT1, diffT, w3,
                                                          u_vec, mlp_w, mlp_b, hT2);
  // depth attention + back to (N,C,V,L)
  final_kernel<<<dim3(VB / 256, NB * LB), 256, 0, stream>>>(hT0, hT1, hT2, w3,
                                                            v_vec, diffT);
  transpose_out_kernel<<<dim3(VB / 256, NB * CB), 256, 0, stream>>>(diffT, out);
}
